// MoE_87428354277803
// MI455X (gfx1250) — matrix operand order repaired and checked by reading, not compiled, not run on hardware
//
#include <hip/hip_runtime.h>
#include <cstdint>

#define NE   8
#define NC   1024
#define NH   2048
#define NB   8
#define NT   2048
#define TOPK 2
#define CAP  640

typedef __attribute__((ext_vector_type(16))) __bf16 v16bf;
typedef __attribute__((ext_vector_type(8)))  __bf16 v8bf;
typedef __attribute__((ext_vector_type(8)))  float  v8f;

union Frag { v16bf v; uint4 q[2]; };

// ---- CDNA5 async global->LDS path (guarded: falls back to reg staging) ----
#if defined(__gfx1250__) && __has_builtin(__builtin_amdgcn_global_load_async_to_lds_b128)
#define HAVE_ASYNC 1
#else
#define HAVE_ASYNC 0
#endif

#if HAVE_ASYNC
typedef int vi4 __attribute__((vector_size(4 * sizeof(int))));
#define GAS __attribute__((address_space(1)))
#define LAS __attribute__((address_space(3)))
__device__ __forceinline__ void async_b128(const __bf16* g, __bf16* l) {
  // per-lane 16B DMA: LDS[l] = MEM[g]; tracked by ASYNCcnt
  __builtin_amdgcn_global_load_async_to_lds_b128(
      (GAS vi4*)(uintptr_t)g, (LAS vi4*)(uint32_t)(uintptr_t)l, 0, 0);
}
__device__ __forceinline__ void wait_async0() {
#if __has_builtin(__builtin_amdgcn_s_wait_asynccnt)
  __builtin_amdgcn_s_wait_asynccnt(0);
#else
  asm volatile("s_wait_asynccnt 0" ::: "memory");
#endif
}
#endif

// ---------------------------------------------------------------- router
__global__ __launch_bounds__(256) void router_kernel(
    const float* __restrict__ x, const float* __restrict__ rw,
    const float* __restrict__ rb, int* __restrict__ tok_e,
    float* __restrict__ tok_p)
{
  int wid  = blockIdx.x * 8 + (threadIdx.x >> 5);   // one wave per token
  int lane = threadIdx.x & 31;
  const float* xp = x + (size_t)wid * NC;
  float acc[NE];
#pragma unroll
  for (int e = 0; e < NE; ++e) acc[e] = 0.f;
  for (int c = lane; c < NC; c += 32) {
    float xv = xp[c];
    const float4* r4 = (const float4*)(rw + (size_t)c * NE);
    float4 a = r4[0], b = r4[1];
    acc[0] += xv * a.x; acc[1] += xv * a.y; acc[2] += xv * a.z; acc[3] += xv * a.w;
    acc[4] += xv * b.x; acc[5] += xv * b.y; acc[6] += xv * b.z; acc[7] += xv * b.w;
  }
#pragma unroll
  for (int e = 0; e < NE; ++e) {
#pragma unroll
    for (int off = 16; off >= 1; off >>= 1)
      acc[e] += __shfl_xor(acc[e], off, 32);
  }
  if (lane == 0) {
    float lg[NE], mx = -1e30f;
#pragma unroll
    for (int e = 0; e < NE; ++e) { lg[e] = acc[e] + rb[e]; mx = fmaxf(mx, lg[e]); }
    float s = 0.f;
#pragma unroll
    for (int e = 0; e < NE; ++e) { lg[e] = __expf(lg[e] - mx); s += lg[e]; }
    float inv = 1.f / s;
    int i0 = 0; float p0v = lg[0];
#pragma unroll
    for (int e = 1; e < NE; ++e) if (lg[e] > p0v) { p0v = lg[e]; i0 = e; }
    int i1 = -1; float p1v = -1.f;
#pragma unroll
    for (int e = 0; e < NE; ++e) if (e != i0 && lg[e] > p1v) { p1v = lg[e]; i1 = e; }
    tok_e[wid * 2 + 0] = i0; tok_e[wid * 2 + 1] = i1;
    tok_p[wid * 2 + 0] = p0v * inv; tok_p[wid * 2 + 1] = p1v * inv;
  }
}

// ---------------------------------------------------------------- dispatch
__global__ void init_slots_kernel(int* __restrict__ slot_token, int n) {
  int i = blockIdx.x * blockDim.x + threadIdx.x;
  if (i < n) slot_token[i] = -1;
}

__global__ void dispatch_kernel(const int* __restrict__ tok_e,
                                int* __restrict__ tok_pos,
                                int* __restrict__ slot_token)
{
  int b = blockIdx.x, lane = threadIdx.x;
  const int NI  = NT * TOPK;   // 4096 routing choices, token-major, k inner
  const int PER = NI / 32;     // 128 per lane, contiguous -> order preserved
  const int* te = tok_e + (size_t)b * NI;
  int cnt[NE];
#pragma unroll
  for (int e = 0; e < NE; ++e) cnt[e] = 0;
  int i0 = lane * PER;
  for (int i = 0; i < PER; ++i) cnt[te[i0 + i]]++;
  int run[NE];
#pragma unroll
  for (int e = 0; e < NE; ++e) {
    int inc = cnt[e];
    for (int off = 1; off < 32; off <<= 1) {
      int u = __shfl_up(inc, off, 32);
      if (lane >= off) inc += u;
    }
    run[e] = inc - cnt[e];     // exclusive prefix = base position
  }
  int* st = slot_token + (size_t)b * NE * CAP;
  int* tp = tok_pos + (size_t)b * NI;
  for (int i = 0; i < PER; ++i) {
    int idx = i0 + i;
    int e = te[idx];
    int pos = run[e]++;
    if (pos < CAP) { st[e * CAP + pos] = idx >> 1; tp[idx] = pos; }
    else           { tp[idx] = -1; }
  }
}

// ---------------------------------------------------------------- gather xe (bf16)
__global__ __launch_bounds__(128) void gather_kernel(
    const float* __restrict__ x, const int* __restrict__ slot_token,
    __bf16* __restrict__ xe)
{
  int row = blockIdx.x;                 // (b*NE + e)*CAP + slot
  int b = row / (NE * CAP);
  int t = slot_token[row];
  __bf16* dst = xe + (size_t)row * NC;
  int c0 = threadIdx.x * 8;
  v8bf out;
  if (t < 0) {
#pragma unroll
    for (int u = 0; u < 8; ++u) out[u] = (__bf16)0.f;
  } else {
    const float* src = x + ((size_t)b * NT + t) * NC + c0;
    float4 a = *(const float4*)(src);
    float4 c = *(const float4*)(src + 4);
    out[0] = (__bf16)a.x; out[1] = (__bf16)a.y; out[2] = (__bf16)a.z; out[3] = (__bf16)a.w;
    out[4] = (__bf16)c.x; out[5] = (__bf16)c.y; out[6] = (__bf16)c.z; out[7] = (__bf16)c.w;
  }
  *(v8bf*)(dst + c0) = out;
}

// --------------------------------------------- fp32 [K][N] -> bf16 [N][K] transpose
__global__ __launch_bounds__(256) void transpose_kernel(
    const float* __restrict__ wg, const float* __restrict__ wf,
    const float* __restrict__ wp,
    __bf16* __restrict__ wgT, __bf16* __restrict__ wfT, __bf16* __restrict__ wpT)
{
  int mat = blockIdx.z, e = blockIdx.y;
  int K = (mat == 2) ? NH : NC;
  int N = (mat == 2) ? NC : NH;
  const float* src = ((mat == 0) ? wg : (mat == 1) ? wf : wp) + (size_t)e * K * N;
  __bf16*      dst = ((mat == 0) ? wgT : (mat == 1) ? wfT : wpT) + (size_t)e * K * N;
  int tilesN = N / 32;
  int tk = ((int)blockIdx.x / tilesN) * 32;
  int tn = ((int)blockIdx.x % tilesN) * 32;
  __shared__ float tile[32][33];
  int tx = threadIdx.x, ty = threadIdx.y;   // block (32,8)
#pragma unroll
  for (int r = 0; r < 32; r += 8)
    tile[ty + r][tx] = src[(size_t)(tk + ty + r) * N + tn + tx];
  __syncthreads();
#pragma unroll
  for (int r = 0; r < 32; r += 8)
    dst[(size_t)(tn + ty + r) * K + tk + tx] = (__bf16)tile[tx][ty + r];
}

// ---------------------------------------------------------------- WMMA GEMM
// A: [gridDim.y][CAP][K] bf16 row-major.  Bt: [NE][N][K] bf16 (pre-transposed).
// FUSED: out = silu(A@B1 + bias1) * (A@B2 + bias2);  else out = A@B1 + bias1.
template <bool FUSED>
__global__ __launch_bounds__(256) void gemm_kernel(
    const __bf16* __restrict__ A, const __bf16* __restrict__ Bt1,
    const __bf16* __restrict__ Bt2, const float* __restrict__ bias1,
    const float* __restrict__ bias2, __bf16* __restrict__ out, int K, int N)
{
  const int tid = threadIdx.x;
  const int e = (int)blockIdx.y & (NE - 1);
  const int tilesN = N / 128;
  const int tm = (int)blockIdx.x / tilesN;   // 64-row tile
  const int tn = (int)blockIdx.x % tilesN;   // 128-col tile

  const __bf16* Ab  = A   + (size_t)blockIdx.y * CAP * K + (size_t)(tm * 64) * K;
  const __bf16* B1b = Bt1 + (size_t)e * N * K + (size_t)(tn * 128) * K;
  const __bf16* B2b = FUSED ? (Bt2 + (size_t)e * N * K + (size_t)(tn * 128) * K) : nullptr;

  // LDS layout (halves): per-buffer [A 64x40 | B1 128x40 | B2 128x40]
  constexpr int NBUFS = HAVE_ASYNC ? 2 : 1;
  constexpr int AH    = 64 * 40;
  constexpr int BH    = 128 * 40;
  constexpr int BUFH  = AH + BH * (FUSED ? 2 : 1);
  constexpr int EPIH  = 64 * 136;            // epilogue restage [64][128+8]
  constexpr int SMEMH = (NBUFS * BUFH > EPIH) ? NBUFS * BUFH : EPIH;
  __shared__ __align__(16) __bf16 smem[SMEMH];
  auto pA  = [&](int b2) { return smem + b2 * BUFH; };
  auto pB1 = [&](int b2) { return smem + b2 * BUFH + AH; };
  auto pB2 = [&](int b2) { return smem + b2 * BUFH + AH + BH; };

  const int am = tid >> 2, ak = (tid & 3) * 8;     // A: one 16B chunk / thread
  const int bn = tid >> 1, bk = (tid & 1) * 16;    // B: one 32B chunk / thread

  const int lane = tid & 31;
  const int half = lane >> 4;                      // which half-wave
  const int lr   = lane & 15;
  const int wv   = tid >> 5;
  const int wm   = (wv & 1) * 32;                  // wave M offset (2x4 wave grid)
  const int wn   = (wv >> 1) * 32;                 // wave N offset

  v8f acc1[2][2], acc2[2][2];
#pragma unroll
  for (int i = 0; i < 2; ++i)
#pragma unroll
    for (int j = 0; j < 2; ++j) { acc1[i][j] = {}; acc2[i][j] = {}; }

#if HAVE_ASYNC
  auto issue_tile = [&](int kk, int b2) {
    async_b128(Ab  + (size_t)am * K + kk + ak,     pA(b2)  + am * 40 + ak);
    async_b128(B1b + (size_t)bn * K + kk + bk,     pB1(b2) + bn * 40 + bk);
    async_b128(B1b + (size_t)bn * K + kk + bk + 8, pB1(b2) + bn * 40 + bk + 8);
    if constexpr (FUSED) {
      async_b128(B2b + (size_t)bn * K + kk + bk,     pB2(b2) + bn * 40 + bk);
      async_b128(B2b + (size_t)bn * K + kk + bk + 8, pB2(b2) + bn * 40 + bk + 8);
    }
  };
  issue_tile(0, 0);
#else
  uint4 ra, rb1a, rb1b, rb2a = {}, rb2b = {};
  ra   = *(const uint4*)(Ab  + (size_t)am * K + ak);
  rb1a = *(const uint4*)(B1b + (size_t)bn * K + bk);
  rb1b = *(const uint4*)(B1b + (size_t)bn * K + bk + 8);
  if constexpr (FUSED) {
    rb2a = *(const uint4*)(B2b + (size_t)bn * K + bk);
    rb2b = *(const uint4*)(B2b + (size_t)bn * K + bk + 8);
  }
#endif

  for (int kk = 0, ib = 0; kk < K; kk += 32, ib ^= (NBUFS - 1)) {
#if HAVE_ASYNC
    wait_async0();                     // buffer ib is resident in LDS
    __syncthreads();                   // ...for every wave; prev buffer free
    if (kk + 32 < K) issue_tile(kk + 32, ib ^ 1);  // DMA next tile under compute
#else
    __syncthreads();
    *(uint4*)(pA(0)  + am * 40 + ak)     = ra;
    *(uint4*)(pB1(0) + bn * 40 + bk)     = rb1a;
    *(uint4*)(pB1(0) + bn * 40 + bk + 8) = rb1b;
    if constexpr (FUSED) {
      *(uint4*)(pB2(0) + bn * 40 + bk)     = rb2a;
      *(uint4*)(pB2(0) + bn * 40 + bk + 8) = rb2b;
    }
    if (kk + 32 < K) {                 // software pipeline next tile via regs
      int k2 = kk + 32;
      ra   = *(const uint4*)(Ab  + (size_t)am * K + k2 + ak);
      rb1a = *(const uint4*)(B1b + (size_t)bn * K + k2 + bk);
      rb1b = *(const uint4*)(B1b + (size_t)bn * K + k2 + bk + 8);
      if constexpr (FUSED) {
        rb2a = *(const uint4*)(B2b + (size_t)bn * K + k2 + bk);
        rb2b = *(const uint4*)(B2b + (size_t)bn * K + k2 + bk + 8);
      }
    }
    __syncthreads();
#endif

    // Fragments (contiguous k order, same for A and B):
    // A 16x32: lane lr row; K halves [half*16 .. +15] (contiguous in A row)
    // B 32x16: lane lr col; K halves [half*16 .. +15] (contiguous in B^T row)
    Frag af[2], bf1[2], bf2[2];
#pragma unroll
    for (int i = 0; i < 2; ++i) {
      const __bf16* r = pA(ib) + (wm + i * 16 + lr) * 40;
      af[i].q[0] = *(const uint4*)(r + half * 16);
      af[i].q[1] = *(const uint4*)(r + half * 16 + 8);
    }
#pragma unroll
    for (int j = 0; j < 2; ++j) {
      const __bf16* r1 = pB1(ib) + (wn + j * 16 + lr) * 40;
      bf1[j].q[0] = *(const uint4*)(r1 + half * 16);
      bf1[j].q[1] = *(const uint4*)(r1 + half * 16 + 8);
      if constexpr (FUSED) {
        const __bf16* r2 = pB2(ib) + (wn + j * 16 + lr) * 40;
        bf2[j].q[0] = *(const uint4*)(r2 + half * 16);
        bf2[j].q[1] = *(const uint4*)(r2 + half * 16 + 8);
      }
    }
#pragma unroll
    for (int i = 0; i < 2; ++i)
#pragma unroll
      for (int j = 0; j < 2; ++j) {
        acc1[i][j] = __builtin_amdgcn_wmma_f32_16x16x32_bf16(
            false, af[i].v, false, bf1[j].v, (short)0, acc1[i][j], false, false);
        if constexpr (FUSED)
          acc2[i][j] = __builtin_amdgcn_wmma_f32_16x16x32_bf16(
              false, af[i].v, false, bf2[j].v, (short)0, acc2[i][j], false, false);
      }
  }

  // ------- epilogue: bias + (silu*mul) -> restage in LDS -> coalesced stores
  __syncthreads();                        // all LDS reads done; reuse smem
  const float* bb1 = bias1 + (size_t)e * N;
  const float* bb2 = FUSED ? (bias2 + (size_t)e * N) : nullptr;
#pragma unroll
  for (int i = 0; i < 2; ++i)
#pragma unroll
    for (int j = 0; j < 2; ++j) {
      int col = wn + j * 16 + lr;         // tile-local column
      float b1 = bb1[tn * 128 + col];
      float b2 = FUSED ? bb2[tn * 128 + col] : 0.f;
#pragma unroll
      for (int r = 0; r < 8; ++r) {
        int row = wm + i * 16 + half * 8 + r;   // tile-local row (C/D layout)
        float v;
        if constexpr (FUSED) {
          float g = acc1[i][j][r] + b1;
          float f = acc2[i][j][r] + b2;
          v = g * (1.f / (1.f + __expf(-g))) * f;   // silu(g) * f
        } else {
          v = acc1[i][j][r] + b1;
        }
        smem[row * 136 + col] = (__bf16)v;
      }
    }
  __syncthreads();
  __bf16* ob = out + (size_t)blockIdx.y * CAP * N + (size_t)(tm * 64) * N + tn * 128;
#pragma unroll
  for (int u = 0; u < 4; ++u) {
    int idx = tid * 4 + u;                // 1024 x 16B covers 64x128 bf16
    int r = idx >> 4, c = (idx & 15) * 8;
    *(uint4*)(ob + (size_t)r * N + c) = *(const uint4*)(smem + r * 136 + c);
  }
}

// ---------------------------------------------------------------- combine
__global__ __launch_bounds__(256) void combine_kernel(
    const __bf16* __restrict__ ye, const int* __restrict__ tok_e,
    const int* __restrict__ tok_pos, const float* __restrict__ tok_p,
    float* __restrict__ y)
{
  int bt = blockIdx.x;
  int b  = bt / NT;
  int e0 = tok_e[bt * 2], e1 = tok_e[bt * 2 + 1];
  int q0 = tok_pos[bt * 2], q1 = tok_pos[bt * 2 + 1];
  float p0 = tok_p[bt * 2], p1 = tok_p[bt * 2 + 1];
  const __bf16* y0 = ye + ((size_t)(b * NE + e0) * CAP + (q0 < 0 ? 0 : q0)) * NC;
  const __bf16* y1 = ye + ((size_t)(b * NE + e1) * CAP + (q1 < 0 ? 0 : q1)) * NC;
  int c = threadIdx.x * 4;
  float r0 = 0.f, r1 = 0.f, r2 = 0.f, r3 = 0.f;
  if (q0 >= 0) {
    r0 += p0 * (float)y0[c];     r1 += p0 * (float)y0[c + 1];
    r2 += p0 * (float)y0[c + 2]; r3 += p0 * (float)y0[c + 3];
  }
  if (q1 >= 0) {
    r0 += p1 * (float)y1[c];     r1 += p1 * (float)y1[c + 1];
    r2 += p1 * (float)y1[c + 2]; r3 += p1 * (float)y1[c + 3];
  }
  float4 o; o.x = r0; o.y = r1; o.z = r2; o.w = r3;
  *(float4*)(y + (size_t)bt * NC + c) = o;
}

// ---------------------------------------------------------------- launch
extern "C" void kernel_launch(void* const* d_in, const int* in_sizes, int n_in,
                              void* d_out, int out_size, void* d_ws, size_t ws_size,
                              hipStream_t stream)
{
  const float* x   = (const float*)d_in[0];
  const float* wfc = (const float*)d_in[1];
  const float* bfc = (const float*)d_in[2];
  const float* wgt = (const float*)d_in[3];
  const float* bgt = (const float*)d_in[4];
  const float* wpj = (const float*)d_in[5];
  const float* bpj = (const float*)d_in[6];
  const float* rw  = (const float*)d_in[7];
  const float* rb  = (const float*)d_in[8];
  float* y = (float*)d_out;

  char* w = (char*)d_ws;
  auto alloc = [&](size_t bytes) -> void* {
    void* p = (void*)w;
    w += (bytes + 255) & ~(size_t)255;
    return p;
  };
  __bf16* xe    = (__bf16*)alloc((size_t)NB * NE * CAP * NC * 2);
  __bf16* hbuf  = (__bf16*)alloc((size_t)NB * NE * CAP * NH * 2);
  __bf16* yebuf = (__bf16*)alloc((size_t)NB * NE * CAP * NC * 2);
  __bf16* wgT   = (__bf16*)alloc((size_t)NE * NH * NC * 2);
  __bf16* wfT   = (__bf16*)alloc((size_t)NE * NH * NC * 2);
  __bf16* wpT   = (__bf16*)alloc((size_t)NE * NC * NH * 2);
  int*   tok_e      = (int*)alloc((size_t)NB * NT * TOPK * 4);
  int*   tok_pos    = (int*)alloc((size_t)NB * NT * TOPK * 4);
  float* tok_p      = (float*)alloc((size_t)NB * NT * TOPK * 4);
  int*   slot_token = (int*)alloc((size_t)NB * NE * CAP * 4);

  router_kernel<<<NB * NT / 8, 256, 0, stream>>>(x, rw, rb, tok_e, tok_p);

  int nslots = NB * NE * CAP;
  init_slots_kernel<<<(nslots + 255) / 256, 256, 0, stream>>>(slot_token, nslots);
  dispatch_kernel<<<NB, 32, 0, stream>>>(tok_e, tok_pos, slot_token);
  gather_kernel<<<nslots, 128, 0, stream>>>(x, slot_token, xe);

  transpose_kernel<<<dim3(2048, NE, 3), dim3(32, 8), 0, stream>>>(
      wgt, wfc, wpj, wgT, wfT, wpT);

  gemm_kernel<true><<<dim3((CAP / 64) * (NH / 128), NB * NE), 256, 0, stream>>>(
      xe, wgT, wfT, bgt, bfc, hbuf, NC, NH);
  gemm_kernel<false><<<dim3((CAP / 64) * (NC / 128), NB * NE), 256, 0, stream>>>(
      hbuf, wpT, nullptr, bpj, nullptr, yebuf, NH, NC);

  combine_kernel<<<NB * NT, 256, 0, stream>>>(yebuf, tok_e, tok_pos, tok_p, y);
}
